// Receiver_61083024884023
// MI455X (gfx1250) — compile-verified, matrix operand order checked by reading, not run on hardware
//
#include <hip/hip_runtime.h>

// GATv2 + message-dot + column softmax for MI455X (gfx1250, wave32).
// Dense hot spot (h @ msg_emb^T) runs on v_wmma_f32_16x16x32_f16.

#define NEG_SLOPE 0.2f

typedef __attribute__((ext_vector_type(16))) _Float16 v16h;
typedef __attribute__((ext_vector_type(8)))  _Float16 v8h;
typedef __attribute__((ext_vector_type(8)))  float    v8f;

// Monotonic float <-> uint encoding so we can use integer atomicMax for
// scatter-max (segment_max) and column-max.
__device__ __forceinline__ unsigned int f2ord(float f) {
  unsigned int u = __float_as_uint(f);
  return (u & 0x80000000u) ? ~u : (u | 0x80000000u);
}
__device__ __forceinline__ float ord2f(unsigned int u) {
  unsigned int b = (u & 0x80000000u) ? (u & 0x7FFFFFFFu) : ~u;
  return __uint_as_float(b);
}

// ---- init scratch accumulators (smax/denom per node-head, col reductions) ----
__global__ void k_init(unsigned int* __restrict__ smax, float* __restrict__ denom,
                       unsigned int* __restrict__ colmax, float* __restrict__ colsum,
                       int n2, int ncol) {
  int i = blockIdx.x * blockDim.x + threadIdx.x;
  if (i < n2)  { smax[i] = 0u; denom[i] = 0.0f; }   // enc(0) < enc(-inf): safe bottom
  if (i < ncol){ colmax[i] = 0u; colsum[i] = 0.0f; }
}

// ---- xl = x@Wl + bl ; xr = x@Wr + br  (F_IN = 2) ----
__global__ void k_node_transform(const float* __restrict__ x,
                                 const float* __restrict__ Wl, const float* __restrict__ bl,
                                 const float* __restrict__ Wr, const float* __restrict__ br,
                                 float* __restrict__ xl, float* __restrict__ xr, int n) {
  int idx = blockIdx.x * blockDim.x + threadIdx.x;
  if (idx >= n * 128) return;
  int node = idx >> 7, ch = idx & 127;
  float x0 = x[node * 2 + 0], x1 = x[node * 2 + 1];
  xl[idx] = fmaf(x0, Wl[ch], fmaf(x1, Wl[128 + ch], bl[ch]));
  xr[idx] = fmaf(x0, Wr[ch], fmaf(x1, Wr[128 + ch], br[ch]));
}

// ---- per-edge attention score + scatter-max over dst ----
__global__ void k_edge_score(const int* __restrict__ ei, const float* __restrict__ eattr,
                             const float* __restrict__ xl, const float* __restrict__ xr,
                             const float* __restrict__ We, const float* __restrict__ att,
                             float* __restrict__ score, unsigned int* __restrict__ smax, int E) {
  int e = blockIdx.x * blockDim.x + threadIdx.x;
  if (e >= E) return;
  int s = ei[e], d = ei[E + e];
  float ea = eattr[e];
  const float* pl = xl + (size_t)s * 128;
  const float* pr = xr + (size_t)d * 128;
  float sc0 = 0.0f, sc1 = 0.0f;
#pragma unroll 8
  for (int ch = 0; ch < 64; ++ch) {
    float m = pl[ch] + pr[ch] + ea * We[ch];
    m = m > 0.0f ? m : NEG_SLOPE * m;
    sc0 = fmaf(m, att[ch], sc0);
  }
#pragma unroll 8
  for (int ch = 64; ch < 128; ++ch) {
    float m = pl[ch] + pr[ch] + ea * We[ch];
    m = m > 0.0f ? m : NEG_SLOPE * m;
    sc1 = fmaf(m, att[ch], sc1);
  }
  score[(size_t)e * 2 + 0] = sc0;
  score[(size_t)e * 2 + 1] = sc1;
  atomicMax(smax + d * 2 + 0, f2ord(sc0));
  atomicMax(smax + d * 2 + 1, f2ord(sc1));
}

// ---- acc = bias (aliases xr; runs after k_edge_score in stream order) ----
__global__ void k_acc_init(float* __restrict__ acc, const float* __restrict__ bias, int n) {
  int idx = blockIdx.x * blockDim.x + threadIdx.x;
  if (idx < n * 128) acc[idx] = bias[idx & 127];
}

// ---- ex = exp(score - smax[dst]) in place; denom[dst] += ex ----
__global__ void k_edge_exp(const int* __restrict__ ei, float* __restrict__ score,
                           const unsigned int* __restrict__ smax, float* __restrict__ denom, int E) {
  int idx = blockIdx.x * blockDim.x + threadIdx.x;
  if (idx >= E * 2) return;
  int e = idx >> 1, h = idx & 1;
  int d = ei[E + e];
  float ex = expf(score[idx] - ord2f(smax[d * 2 + h]));
  score[idx] = ex;
  atomicAdd(denom + d * 2 + h, ex);
}

// ---- out[dst] += xl[src] * alpha   (alpha = ex / (denom + 1e-16)) ----
__global__ void k_aggregate(const int* __restrict__ ei, const float* __restrict__ ex,
                            const float* __restrict__ denom, const float* __restrict__ xl,
                            float* __restrict__ acc, int E) {
  size_t idx = (size_t)blockIdx.x * blockDim.x + threadIdx.x;
  if (idx >= (size_t)E * 128) return;
  int e = (int)(idx >> 7), ch = (int)(idx & 127);
  int s = ei[e], d = ei[E + e];
  int h = ch >> 6;
  float a = ex[(size_t)e * 2 + h] / (denom[d * 2 + h] + 1e-16f);
  atomicAdd(acc + (size_t)d * 128 + ch, xl[(size_t)s * 128 + ch] * a);
}

// ---- h = relu(acc) -> f16 for WMMA ----
__global__ void k_relu_f16(const float* __restrict__ acc, _Float16* __restrict__ hf, int n) {
  int idx = blockIdx.x * blockDim.x + threadIdx.x;
  if (idx < n * 128) hf[idx] = (_Float16)fmaxf(acc[idx], 0.0f);
}

// ---- msg_emb = message @ Wfc + bfc  -> f16 (only B rows; concat dup is mirrored later) ----
__global__ void k_msg_emb(const float* __restrict__ msg, const float* __restrict__ Wfc,
                          const float* __restrict__ bfc, _Float16* __restrict__ me, int B) {
  int idx = blockIdx.x * blockDim.x + threadIdx.x;
  if (idx >= B * 128) return;
  int b = idx >> 7, ch = idx & 127;
  float s = bfc[ch];
  const float* mrow = msg + (size_t)b * 128;
#pragma unroll 8
  for (int k = 0; k < 128; ++k) s = fmaf(mrow[k], Wfc[k * 128 + ch], s);
  me[idx] = (_Float16)s;
}

// ---- dot[:, 0:B] = h @ msg_emb^T via v_wmma_f32_16x16x32_f16, K=128 in 4 steps ----
// One 16x16 tile per wave; 8 waves per block. Layouts per CDNA5 ISA 7.12.2:
//   A lane (m = lane&15, half = lane>>4): halves [k0+8*half .. +7] and [k0+16+8*half .. +7]
//   B is symmetric with column n = lane&15 (row n of msg_emb == column n of B).
__global__ __launch_bounds__(256) void k_gemm_wmma(const _Float16* __restrict__ hf,
                                                   const _Float16* __restrict__ me,
                                                   float* __restrict__ out,
                                                   int Mtiles, int outw) {
  int lane = threadIdx.x & 31;
  int wave = threadIdx.x >> 5;
  int tileN = blockIdx.x;
  int tileM = blockIdx.y * 8 + wave;
  if (tileM >= Mtiles) return;              // wave-uniform: EXEC all-ones for WMMA
  int half = lane >> 4;
  int r = lane & 15;
  const _Float16* arow = hf + (size_t)(tileM * 16 + r) * 128;
  const _Float16* brow = me + (size_t)(tileN * 16 + r) * 128;
  v8f acc = {};
#pragma unroll
  for (int kk = 0; kk < 4; ++kk) {
    int kb = kk * 32 + half * 8;
    v8h a0 = *(const v8h*)(arow + kb);
    v8h a1 = *(const v8h*)(arow + kb + 16);
    v8h b0 = *(const v8h*)(brow + kb);
    v8h b1 = *(const v8h*)(brow + kb + 16);
    v16h A, Bv;
#pragma unroll
    for (int i = 0; i < 8; ++i) { A[i] = a0[i]; A[8 + i] = a1[i]; Bv[i] = b0[i]; Bv[8 + i] = b1[i]; }
    acc = __builtin_amdgcn_wmma_f32_16x16x32_f16(false, A, false, Bv,
                                                 (short)0, acc, false, false);
  }
  int col = tileN * 16 + r;                  // N = lane&15
#pragma unroll
  for (int rr = 0; rr < 8; ++rr) {           // M = rr + 8*half
    int row = tileM * 16 + rr + 8 * half;
    out[(size_t)row * outw + col] = acc[rr];
  }
}

// ---- column max over rows (cols 0..B-1) ----
__global__ void k_colmax(const float* __restrict__ out, unsigned int* __restrict__ colmax,
                         int rows, int ncol, int outw, int chunk) {
  int col = blockIdx.x * blockDim.x + threadIdx.x;
  if (col >= ncol) return;
  int r0 = blockIdx.y * chunk;
  int r1 = r0 + chunk; if (r1 > rows) r1 = rows;
  float m = -3.4e38f;
  for (int r = r0; r < r1; ++r) m = fmaxf(m, out[(size_t)r * outw + col]);
  atomicMax(colmax + col, f2ord(m));
}

// ---- column sum of exp(x - max) ----
__global__ void k_colsum(const float* __restrict__ out, const unsigned int* __restrict__ colmax,
                         float* __restrict__ colsum, int rows, int ncol, int outw, int chunk) {
  int col = blockIdx.x * blockDim.x + threadIdx.x;
  if (col >= ncol) return;
  int r0 = blockIdx.y * chunk;
  int r1 = r0 + chunk; if (r1 > rows) r1 = rows;
  float mx = ord2f(colmax[col]);
  float s = 0.0f;
  for (int r = r0; r < r1; ++r) s += expf(out[(size_t)r * outw + col] - mx);
  atomicAdd(colsum + col, s);
}

// ---- normalize and mirror into the duplicated column half ----
__global__ void k_normalize(float* __restrict__ out, const unsigned int* __restrict__ colmax,
                            const float* __restrict__ colsum, int rows, int ncol, int outw) {
  size_t idx = (size_t)blockIdx.x * blockDim.x + threadIdx.x;
  if (idx >= (size_t)rows * ncol) return;
  int col = (int)(idx % (unsigned)ncol);
  int row = (int)(idx / (unsigned)ncol);
  float v = expf(out[(size_t)row * outw + col] - ord2f(colmax[col])) / colsum[col];
  out[(size_t)row * outw + col] = v;
  out[(size_t)row * outw + col + ncol] = v;   // dot[:, j+B] == dot[:, j]
}

extern "C" void kernel_launch(void* const* d_in, const int* in_sizes, int n_in,
                              void* d_out, int out_size, void* d_ws, size_t ws_size,
                              hipStream_t stream) {
  const float* message = (const float*)d_in[0];
  const float* x       = (const float*)d_in[1];
  const int*   ei      = (const int*)d_in[2];
  const float* eattr   = (const float*)d_in[3];
  // conv1 params (d_in[4..10]) are dead: h1 never reaches the output.
  const float* Wl2  = (const float*)d_in[11];
  const float* bl2  = (const float*)d_in[12];
  const float* Wr2  = (const float*)d_in[13];
  const float* br2  = (const float*)d_in[14];
  const float* We2  = (const float*)d_in[15];
  const float* att2 = (const float*)d_in[16];
  const float* bias2= (const float*)d_in[17];
  const float* Wfc  = (const float*)d_in[18];
  const float* bfc  = (const float*)d_in[19];
  float* out = (float*)d_out;

  const int N = in_sizes[1] / 2;       // 50000 (divisible by 16)
  const int E = in_sizes[2] / 2;       // 800000
  const int B = in_sizes[0] / 128;     // 512
  const int OUTW = 2 * B;              // 1024

  // workspace layout (~71 MB); xr is reused as the aggregation accumulator.
  char* ws = (char*)d_ws;
  size_t o = 0;
  float*        xl     = (float*)(ws + o);        o += (size_t)N * 128 * 4;
  float*        xr     = (float*)(ws + o);        o += (size_t)N * 128 * 4;
  float*        acc    = xr;
  float*        score  = (float*)(ws + o);        o += (size_t)E * 2 * 4;
  unsigned int* smax   = (unsigned int*)(ws + o); o += (size_t)N * 2 * 4;
  float*        denom  = (float*)(ws + o);        o += (size_t)N * 2 * 4;
  _Float16*     hf     = (_Float16*)(ws + o);     o += (size_t)N * 128 * 2;
  _Float16*     me     = (_Float16*)(ws + o);     o += (size_t)B * 128 * 2;
  unsigned int* colmax = (unsigned int*)(ws + o); o += (size_t)B * 4;
  float*        colsum = (float*)(ws + o);

  const int T = 256;
  int initN = (2 * N > B) ? 2 * N : B;
  k_init<<<(initN + T - 1) / T, T, 0, stream>>>(smax, denom, colmax, colsum, 2 * N, B);
  k_node_transform<<<(N * 128 + T - 1) / T, T, 0, stream>>>(x, Wl2, bl2, Wr2, br2, xl, xr, N);
  k_edge_score<<<(E + T - 1) / T, T, 0, stream>>>(ei, eattr, xl, xr, We2, att2, score, smax, E);
  k_acc_init<<<(N * 128 + T - 1) / T, T, 0, stream>>>(acc, bias2, N);
  k_edge_exp<<<(2 * E + T - 1) / T, T, 0, stream>>>(ei, score, smax, denom, E);
  {
    size_t tot = (size_t)E * 128;
    k_aggregate<<<(unsigned)((tot + T - 1) / T), T, 0, stream>>>(ei, score, denom, xl, acc, E);
  }
  k_relu_f16<<<(N * 128 + T - 1) / T, T, 0, stream>>>(acc, hf, N);
  k_msg_emb<<<(B * 128 + T - 1) / T, T, 0, stream>>>(message, Wfc, bfc, me, B);

  int Mtiles = N / 16;                 // 3125
  dim3 gg(B / 16, (Mtiles + 7) / 8);   // 32 x 391, 8 waves/block
  k_gemm_wmma<<<gg, 256, 0, stream>>>(hf, me, out, Mtiles, OUTW);

  const int CHUNK = 256;
  dim3 gm((B + T - 1) / T, (N + CHUNK - 1) / CHUNK);
  k_colmax<<<gm, T, 0, stream>>>(out, colmax, N, B, OUTW, CHUNK);
  k_colsum<<<gm, T, 0, stream>>>(out, colmax, colsum, N, B, OUTW, CHUNK);
  {
    size_t tot = (size_t)N * B;
    k_normalize<<<(unsigned)((tot + T - 1) / T), T, 0, stream>>>(out, colmax, colsum, N, B, OUTW);
  }
}
